// CoAttn_2439541424737
// MI455X (gfx1250) — matrix operand order repaired and checked by reading, not compiled, not run on hardware
//
#include <hip/hip_runtime.h>

#define MIN_VAL  -99999999.0f
#define SCALE    0.0625f            // 1/sqrt(256)

typedef __attribute__((ext_vector_type(16))) __bf16 v16bf;
typedef __attribute__((ext_vector_type(8)))  __bf16 v8bf;
typedef __attribute__((ext_vector_type(8)))  float  v8f;

// ---------------------------------------------------------------------------
// Fragment layouts used here (wave32; A/B use contiguous k, not the ISA table order):
// A 16x32 bf16 : lane m=l&15 is row M; elem e -> K = e + 16*(l>=16)
// B 32x16 bf16 : lane n=l&15 is col N; elem e -> K = e + 16*(l>=16)
// C 16x16 f32  : vgpr r, lane l -> M = r + 8*(l>=16), N = l&15
// ---------------------------------------------------------------------------

__device__ __forceinline__ v16bf ldA_bf(const __bf16* __restrict__ arow, int kc) {
  v8bf lo = *(const v8bf*)(arow + kc);
  v8bf hi = *(const v8bf*)(arow + kc + 8);
  return __builtin_shufflevector(lo, hi, 0,1,2,3,4,5,6,7,8,9,10,11,12,13,14,15);
}

__device__ __forceinline__ v16bf ldB_bf(const __bf16* __restrict__ brow, int kc) {
  return *(const v16bf*)(brow + kc);
}

__device__ __forceinline__ v16bf ldA_lds(const __bf16* __restrict__ row) {
  v8bf lo = *(const v8bf*)(row);
  v8bf hi = *(const v8bf*)(row + 8);
  return __builtin_shufflevector(lo, hi, 0,1,2,3,4,5,6,7,8,9,10,11,12,13,14,15);
}

__device__ __forceinline__ v8f wmma_bf16(v16bf a, v16bf b, v8f c) {
  return __builtin_amdgcn_wmma_f32_16x16x32_bf16(false, a, false, b, (short)0, c,
                                                 false, false);
}

__device__ __forceinline__ v8f vzero8() {
  v8f z;
#pragma unroll
  for (int i = 0; i < 8; ++i) z[i] = 0.0f;
  return z;
}

// one-exp online softmax update: (rm, rs) <- merge with single value v
__device__ __forceinline__ void smax_upd(float& rm, float& rs, float v) {
  float d = v - rm;
  float t = __expf(-fabsf(d));
  if (d > 0.0f) { rs = rs * t + 1.0f; rm = v; }
  else          { rs = rs + t; }
}

// one-exp merge of two (max, sum) pairs
__device__ __forceinline__ void smax_merge(float& mx, float& sm, float om, float os) {
  float d = om - mx;
  float t = __expf(-fabsf(d));
  if (d > 0.0f) { sm = sm * t + os; mx = om; }
  else          { sm = sm + os * t; }
}

// ---------------------------------------------------------------------------
// Kernel 0: one-time f32 -> bf16 conversion (vectorized, 8 elems/thread)
// ---------------------------------------------------------------------------
__global__ void __launch_bounds__(256) k_cvt_bf16(const float* __restrict__ src,
                                                  __bf16* __restrict__ dst, long n8) {
  long i = (long)blockIdx.x * blockDim.x + threadIdx.x;
  if (i >= n8) return;
  const float4* p = (const float4*)(src + i * 8);
  float4 a = p[0], b = p[1];
  v8bf o;
  o[0]=(__bf16)a.x; o[1]=(__bf16)a.y; o[2]=(__bf16)a.z; o[3]=(__bf16)a.w;
  o[4]=(__bf16)b.x; o[5]=(__bf16)b.y; o[6]=(__bf16)b.z; o[7]=(__bf16)b.w;
  *(v8bf*)(dst + i * 8) = o;
}

// ---------------------------------------------------------------------------
// Kernel 1: online softmax stats (max & sumexp) over the N axis of
//           S[M,N] = SCALE * A[M,:] . B[N,:]   (masked with MIN_VAL)
// All structural parameters are compile-time constants.
// ---------------------------------------------------------------------------
template<int AST, int ASH, int BST, int BSH, int MACOLS, int MBCOLS,
         int OCOLS, int MTSH, int MTMASK, int NCT>
__global__ void __launch_bounds__(128) k_stats(
    const __bf16* __restrict__ A, const __bf16* __restrict__ B,
    const int* __restrict__ mA, const int* __restrict__ mB,
    float* __restrict__ oMax, float* __restrict__ oSum)
{
  const int lane = threadIdx.x & 31;
  const int wave = blockIdx.x * (blockDim.x >> 5) + (threadIdx.x >> 5);
  const int b    = wave >> MTSH;
  const int m0   = (wave & MTMASK) << 4;
  const int m    = lane & 15;
  const int g    = (lane >> 4) & 1;

  const __bf16* arow = A + (long)(m0 + m) * AST + (long)(b >> ASH) * 256 + 16 * g;
  v16bf af[8];
#pragma unroll
  for (int kc = 0; kc < 8; ++kc) af[kc] = ldA_bf(arow, kc * 32);

  int mrow[8];
#pragma unroll
  for (int r = 0; r < 8; ++r)
    mrow[r] = mA[(b >> ASH) * MACOLS + m0 + r + 8 * g];

  float rm[8], rs[8];
#pragma unroll
  for (int r = 0; r < 8; ++r) { rm[r] = MIN_VAL; rs[r] = 0.0f; }

  const __bf16* Bb  = B  + (long)(b >> BSH) * 256 + 16 * g;
  const int*    mBb = mB + (b >> BSH) * MBCOLS;

#pragma unroll 1
  for (int j = 0; j < NCT; ++j) {
    const int col = j * 16 + m;
    const __bf16* brow = Bb + (long)col * BST;
    v8f acc = vzero8();
#pragma unroll
    for (int kc = 0; kc < 8; ++kc)
      acc = wmma_bf16(af[kc], ldB_bf(brow, kc * 32), acc);
    const int mc = mBb[col];
#pragma unroll
    for (int r = 0; r < 8; ++r) {
      float v = ((mrow[r] | mc) == 0) ? acc[r] * SCALE : MIN_VAL;
      smax_upd(rm[r], rs[r], v);
    }
  }

  // combine across the 16 lanes sharing each M row (xor stays inside half-wave)
#pragma unroll
  for (int r = 0; r < 8; ++r) {
    float mx = rm[r], sm = rs[r];
#pragma unroll
    for (int off = 1; off < 16; off <<= 1) {
      float om = __shfl_xor(mx, off, 32);
      float os = __shfl_xor(sm, off, 32);
      smax_merge(mx, sm, om, os);
    }
    rm[r] = mx; rs[r] = sm;
  }

  if (m == 0) {
#pragma unroll
    for (int r = 0; r < 8; ++r) {
      long o = (long)b * OCOLS + m0 + r + 8 * g;
      oMax[o] = rm[r];
      oSum[o] = rs[r];
    }
  }
}

// ---------------------------------------------------------------------------
// Kernel 2: Out[M, 0:256] = sum_k P[M,k] * V[k, 0:256]
//   P[M,k] = valid ? exp(SCALE*S[M,k] - stMax[M]) / stSum[M] : 0
//   Scores recomputed with WMMA from bf16 operands; P relayout via per-wave
//   LDS buffer; V chunk (32x256) staged cooperatively per block into LDS.
// ---------------------------------------------------------------------------
template<int AST, int ASH, int BST, int BSH, int MMCOLS, int MKCOLS, int STCOLS,
         int VST, int VSH, int VBF, int MTSH, int MTMASK, int NK>
__global__ void __launch_bounds__(128) k_attn_av(
    const __bf16* __restrict__ A, const __bf16* __restrict__ B,
    const int* __restrict__ mM, const int* __restrict__ mK,
    const float* __restrict__ stMax, const float* __restrict__ stSum,
    const __bf16* __restrict__ Vbf, const float* __restrict__ Vf32,
    float* __restrict__ Out)
{
  __shared__ __align__(16) __bf16 pb[4][16][32];     //  4 KB: per-wave P tiles
  __shared__ __align__(16) __bf16 vbuf[32][264];     // 17 KB: shared V chunk

  const int lane  = threadIdx.x & 31;
  const int wslot = threadIdx.x >> 5;
  const int wave  = blockIdx.x * (blockDim.x >> 5) + wslot;
  const int b     = wave >> MTSH;
  const int m0    = (wave & MTMASK) << 4;
  const int m     = lane & 15;
  const int g     = (lane >> 4) & 1;
  __bf16 (*P)[32] = pb[wslot];

  const __bf16* arow = A + (long)(m0 + m) * AST + (long)(b >> ASH) * 256 + 16 * g;

  float cm[8], ci[8]; int mr[8];
#pragma unroll
  for (int r = 0; r < 8; ++r) {
    int idx = m0 + r + 8 * g;
    cm[r] = stMax[(long)b * STCOLS + idx];
    ci[r] = 1.0f / stSum[(long)b * STCOLS + idx];
    mr[r] = mM[(b >> ASH) * MMCOLS + idx];
  }

  const __bf16* Bb  = B  + (long)(b >> BSH) * 256 + 16 * g;
  const int*    mKb = mK + (b >> BSH) * MKCOLS;
  const long    vBatch = (long)(b >> VSH) * 256;

  v8f acc_o[16];
#pragma unroll
  for (int dt = 0; dt < 16; ++dt) acc_o[dt] = vzero8();

#pragma unroll 1
  for (int kt = 0; kt < NK; ++kt) {
    const int k0 = kt * 32;

    // --- cooperative staging of V[k0:k0+32, 0:256] into LDS (bf16) ---
#pragma unroll
    for (int it = 0; it < 8; ++it) {
      const int i   = threadIdx.x + it * 128;   // 1024 chunks of 8 elems
      const int row = i >> 5;
      const int col = (i & 31) << 3;
      v8bf x;
      if (VBF) {
        const __bf16* pv = Vbf + (long)(k0 + row) * VST + vBatch + col;
        x = *(const v8bf*)pv;
        if (kt + 1 < NK) __builtin_prefetch(pv + 32 * VST, 0, 0);
      } else {
        const float* pf = Vf32 + (long)(k0 + row) * VST + vBatch + col;
        float4 f0 = *(const float4*)pf;
        float4 f1 = *(const float4*)(pf + 4);
        if (kt + 1 < NK) __builtin_prefetch(pf + 32 * VST, 0, 0);
        x[0]=(__bf16)f0.x; x[1]=(__bf16)f0.y; x[2]=(__bf16)f0.z; x[3]=(__bf16)f0.w;
        x[4]=(__bf16)f1.x; x[5]=(__bf16)f1.y; x[6]=(__bf16)f1.z; x[7]=(__bf16)f1.w;
      }
      *(v8bf*)&vbuf[row][col] = x;
    }

    // --- scores for the 16 x 32 strip, converted to probabilities into LDS ---
#pragma unroll
    for (int half = 0; half < 2; ++half) {
      const int kcol = k0 + half * 16 + m;
      const __bf16* brow = Bb + (long)kcol * BST;
      v8f acc = vzero8();
#pragma unroll
      for (int kc = 0; kc < 8; ++kc)
        acc = wmma_bf16(ldA_bf(arow, kc * 32), ldB_bf(brow, kc * 32), acc);
      const int mc = mKb[kcol];
#pragma unroll
      for (int r = 0; r < 8; ++r) {
        float p = 0.0f;
        if ((mr[r] | mc) == 0) p = __expf(acc[r] * SCALE - cm[r]) * ci[r];
        P[r + 8 * g][half * 16 + m] = (__bf16)p;
      }
    }
    __syncthreads();   // vbuf + P stores visible before fragment reads

    // --- P tile as A fragment; accumulate over all 16 d tiles from LDS V ---
    const v16bf pa = ldA_lds(&P[m][16 * g]);
#pragma unroll
    for (int dt = 0; dt < 16; ++dt) {
      v16bf vv;
#pragma unroll
      for (int e = 0; e < 16; ++e) vv[e] = vbuf[e + 16 * g][dt * 16 + m];
      acc_o[dt] = wmma_bf16(pa, vv, acc_o[dt]);
    }
    __syncthreads();   // all waves done reading vbuf before next staging
  }

  float* Ob = Out + (long)b * 256 + m;
#pragma unroll
  for (int dt = 0; dt < 16; ++dt)
#pragma unroll
    for (int r = 0; r < 8; ++r)
      Ob[(long)(m0 + r + 8 * g) * 32768 + dt * 16] = acc_o[dt][r];
}

// ---------------------------------------------------------------------------
// Host side. d_ws layout (bytes):
//   [0, 1572864)          softmax stats (rowmax/rowsum/colmax/colsum, f32)
//   [1572864, 3670016)    q as bf16   (1,048,576 elems)
//   [3670016, 37224448)   node as bf16 (16,777,216 elems)
// d_out = [output_ss (512*128*256) | output_sq (1024*128*256) | cs_input].
// ---------------------------------------------------------------------------
extern "C" void kernel_launch(void* const* d_in, const int* in_sizes, int n_in,
                              void* d_out, int out_size, void* d_ws, size_t ws_size,
                              hipStream_t stream) {
  const float* q     = (const float*)d_in[0];   // [1024, 4, 256]
  const float* node  = (const float*)d_in[1];   // [512, 128, 256]
  const int*   maskq = (const int*)d_in[2];     // [4, 1024]
  const int*   maskn = (const int*)d_in[3];     // [128, 512]

  float* out_ss = (float*)d_out;
  float* out_sq = out_ss + (long)512 * 128 * 256;
  float* out_cs = out_sq + (long)1024 * 128 * 256;

  float* ws     = (float*)d_ws;
  float* rowmax = ws;                 // [128*512]
  float* rowsum = rowmax + 65536;
  float* colmax = rowsum + 65536;     // [128*1024]
  float* colsum = colmax + 131072;
  __bf16* qbf    = (__bf16*)((char*)d_ws + 1572864);
  __bf16* nodebf = (__bf16*)((char*)d_ws + 3670016);

  dim3 blk(128);

  // K0: one-time f32 -> bf16 of both operands
  k_cvt_bf16<<<512, 256, 0, stream>>>(q, qbf, 131072L);
  k_cvt_bf16<<<8192, 256, 0, stream>>>(node, nodebf, 2097152L);

  // K1: per-(b,s2) stats over s1 (for wts1).  A=node, B=q.
  k_stats<32768, 0, 1024, 5, 512, 1024, 512, 5, 31, 64>
      <<<1024, blk, 0, stream>>>(nodebf, qbf, maskn, maskq, rowmax, rowsum);
  // K2: per-(b,s1) stats over s2 (for wts2).  A=q, B=node.
  k_stats<1024, 5, 32768, 0, 1024, 512, 1024, 6, 63, 32>
      <<<2048, blk, 0, stream>>>(qbf, nodebf, maskq, maskn, colmax, colsum);
  // K3: output_sq[t,b,:] = sum_s wts2[b,t,s] * node[s,b,:]
  k_attn_av<1024, 5, 32768, 0, 1024, 512, 1024, 32768, 0, 1, 6, 63, 16>
      <<<2048, blk, 0, stream>>>(qbf, nodebf, maskq, maskn, colmax, colsum,
                                 nodebf, (const float*)nullptr, out_sq);
  // K4a: output_ss[s,b,:] = sum_t wts1[b,s,t] * q[t,b>>5,:]
  k_attn_av<32768, 0, 1024, 5, 512, 1024, 512, 1024, 5, 1, 5, 31, 32>
      <<<1024, blk, 0, stream>>>(nodebf, qbf, maskn, maskq, rowmax, rowsum,
                                 qbf, (const float*)nullptr, out_ss);
  // K4b: cs_input[s,b,:] = sum_t wts1[b,s,t] * output_sq[t,b,:]  (V is f32)
  k_attn_av<32768, 0, 1024, 5, 512, 1024, 512, 32768, 0, 0, 5, 31, 32>
      <<<1024, blk, 0, stream>>>(nodebf, qbf, maskn, maskq, rowmax, rowsum,
                                 (const __bf16*)nullptr, out_sq, out_cs);
}
